// reduce_PAM_Module_36301063586299
// MI455X (gfx1250) — matrix operand order repaired and checked by reading, not compiled, not run on hardware
//
#include <hip/hip_runtime.h>
#include <hip/hip_bf16.h>
#include <cstdint>

#define BB   2
#define CC   512
#define NN   9216
#define KK   64
#define VV   256
#define EPSF 1e-5f

typedef __attribute__((ext_vector_type(16))) __bf16 v16bf;
typedef __attribute__((ext_vector_type(8)))  float  v8f;

union Frag16 { uint4 q[2]; v16bf v; };

static __device__ __forceinline__ unsigned short f2bf(float f) {
  // round-to-nearest-even fp32 -> bf16 (bit container only; no __bf16 math)
  unsigned int u = __builtin_bit_cast(unsigned int, f);
  unsigned int r = u + 0x7fffu + ((u >> 16) & 1u);
  return (unsigned short)(r >> 16);
}

// ---------------------------------------------------------------------------
// Kernel 1: q = BN(Wq x), k = BN(Wk x), v = Wv x + bv   (all -> bf16 workspace)
// q/k layout: [B][N][64]  (row-major per position, for A/B fragment loads)
// v   layout: [B][V][N]   (v-major, so attention LDS tiles need no transpose)
// ---------------------------------------------------------------------------
__global__ __launch_bounds__(256) void pam_proj_qkv(
    const float* __restrict__ x,
    const float* __restrict__ Wq, const float* __restrict__ gq, const float* __restrict__ bq,
    const float* __restrict__ mq, const float* __restrict__ vq,
    const float* __restrict__ Wk, const float* __restrict__ gk, const float* __restrict__ bk,
    const float* __restrict__ mk, const float* __restrict__ vk,
    const float* __restrict__ Wv, const float* __restrict__ bv,
    unsigned short* __restrict__ qws, unsigned short* __restrict__ kws,
    unsigned short* __restrict__ vws)
{
  __shared__ float xs[CC * 32];                       // x tile [c][n], 64 KB
  const int blk = blockIdx.x;
  const int b   = blk / (NN / 32);
  const int n0  = (blk % (NN / 32)) * 32;
  const int t   = threadIdx.x;

  #pragma unroll 4
  for (int r = 0; r < 64; ++r) {                     // stage 512x32 f32, coalesced over n
    int idx = t + 256 * r;
    int c = idx >> 5, n = idx & 31;
    xs[idx] = x[((size_t)b * CC + c) * NN + n0 + n];
  }
  __syncthreads();

  const int n  = t & 31;
  const int cg = t >> 5;                             // 0..7, uniform per wave
  const size_t rowoff = (size_t)(b * NN) + n0 + n;

  for (int g = 0; g < 6; ++g) {
    const int base = cg * 48 + g * 8;                // multiple of 8; never straddles 64/128
    float acc[8] = {0.f,0.f,0.f,0.f,0.f,0.f,0.f,0.f};
    const float* wr;
    if (base < 64)       wr = Wq + (size_t)base * CC;
    else if (base < 128) wr = Wk + (size_t)(base - 64) * CC;
    else                 wr = Wv + (size_t)(base - 128) * CC;

    for (int c = 0; c < CC; ++c) {
      float xv = xs[c * 32 + n];
      #pragma unroll
      for (int u = 0; u < 8; ++u) acc[u] += xv * wr[(size_t)u * CC + c];
    }

    if (base < 64) {
      #pragma unroll
      for (int u = 0; u < 8; ++u) {
        int ch = base + u;
        float s  = gq[ch] * rsqrtf(vq[ch] + EPSF);
        float sh = bq[ch] - mq[ch] * s;
        qws[rowoff * KK + ch] = f2bf(acc[u] * s + sh);
      }
    } else if (base < 128) {
      #pragma unroll
      for (int u = 0; u < 8; ++u) {
        int ch = base - 64 + u;
        float s  = gk[ch] * rsqrtf(vk[ch] + EPSF);
        float sh = bk[ch] - mk[ch] * s;
        kws[rowoff * KK + ch] = f2bf(acc[u] * s + sh);
      }
    } else {
      #pragma unroll
      for (int u = 0; u < 8; ++u) {
        int ch = base - 128 + u;
        // v-major layout: [B][V][N]
        vws[((size_t)(b * VV) + ch) * NN + n0 + n] = f2bf(acc[u] + bv[ch]);
      }
    }
  }
}

// ---------------------------------------------------------------------------
// Kernel 2: fused flash attention. Block = 16 query rows, 4 waves.
// Waves 1-3 stage the V tile (batched b128 copy at constant strides) while
// wave 0 concurrently computes S^T = K_tile x Q (bf16 WMMA, f32 acc) + online
// softmax and publishes P / alpha via LDS. Then all 4 waves do
// O = alpha*O + P x V (each wave owns 64 of the 256 V columns).
// ---------------------------------------------------------------------------
__global__ __launch_bounds__(128) void pam_attn_kernel(
    const unsigned short* __restrict__ qw,
    const unsigned short* __restrict__ kw,
    const unsigned short* __restrict__ vw,   // [B][V][N]
    float* __restrict__ ow)
{
  __shared__ alignas(16) unsigned short sh_p[16 * 32];   // P^T [n][m], bf16
  __shared__ alignas(16) float sh_alpha[16];
  __shared__ alignas(16) float sh_l[16];
  __shared__ alignas(16) unsigned short sh_v[VV * 32];   // V^T [v][m], bf16, 16 KB

  const int qt   = blockIdx.x % (NN / 16);
  const int b    = blockIdx.x / (NN / 16);
  const int tid  = threadIdx.x;
  const int wid  = tid >> 5;
  const int lane = tid & 31;
  const int nloc = lane & 15;
  const int hi     = (lane >= 16);
  const int halfA  = hi ? 8 : 0;    // A-frag / C-row half offset
  const int half16 = hi ? 16 : 0;   // unused (B-frags use halfA too)

  v8f o0 = {0,0,0,0,0,0,0,0}, o1 = {0,0,0,0,0,0,0,0};
  v8f o2 = {0,0,0,0,0,0,0,0}, o3 = {0,0,0,0,0,0,0,0};
  float rmax = -3.0e38f, rsum = 0.0f;

  // ---- hoisted per-thread bases ----
  // waves 1-3: thread t3 copies chunks t3 + 96*r; chunk stride 96 == +24 v rows,
  // so both global and LDS addresses advance by compile-time-constant strides.
  const int t3 = tid - 32;                                         // 0..95 (waves 1-3)
  const unsigned short* gvbase = vw +
      ((size_t)(b * VV) + (t3 >> 2)) * NN + (t3 & 3) * 8;          // + m0 each iter
  unsigned short* lvbase = &sh_v[(t3 >> 2) * 32 + (t3 & 3) * 8];
  const bool tail = (t3 < 64);                                     // wave-uniform

  // wave 0: K row base (advance by 32*KK per iteration)
  const unsigned short* kbase = kw + ((size_t)(b * NN) + nloc) * KK + halfA;

  Frag16 qb0, qb1;                  // Q B-fragments (wave0 only), held all loop
  if (wid == 0) {
    const unsigned short* qp = qw + ((size_t)(b * NN) + qt * 16 + nloc) * KK;
    qb0.q[0] = *(const uint4*)(qp + 0  + halfA);
    qb0.q[1] = *(const uint4*)(qp + 16 + halfA);
    qb1.q[0] = *(const uint4*)(qp + 32 + halfA);
    qb1.q[1] = *(const uint4*)(qp + 48 + halfA);
  }

  for (int it = 0; it < NN / 32; ++it) {
    const int m0 = it * 32;

    if (wid != 0) {
      // --- waves 1-3: stage V tile [256 v][32 m]: load-all then store-all ---
      const unsigned short* gsrc = gvbase + m0;
      uint4 d[10];
      #pragma unroll
      for (int r = 0; r < 10; ++r)
        d[r] = *(const uint4*)(gsrc + (size_t)r * (24 * NN));
      uint4 dt;
      if (tail) dt = *(const uint4*)(gsrc + (size_t)10 * (24 * NN));
      #pragma unroll
      for (int r = 0; r < 10; ++r)
        *(uint4*)(lvbase + r * (24 * 32)) = d[r];
      if (tail) *(uint4*)(lvbase + 10 * (24 * 32)) = dt;
    } else {
      // --- wave 0: S^T (32 m rows x 16 n cols) and online softmax ---
      const unsigned short* kp  = kbase + (size_t)m0 * KK;
      const unsigned short* kp1 = kp + 16 * KK;
      Frag16 a00, a01, a10, a11;
      a00.q[0] = *(const uint4*)(kp);       a00.q[1] = *(const uint4*)(kp + 16);
      a01.q[0] = *(const uint4*)(kp + 32);  a01.q[1] = *(const uint4*)(kp + 48);
      a10.q[0] = *(const uint4*)(kp1);      a10.q[1] = *(const uint4*)(kp1 + 16);
      a11.q[0] = *(const uint4*)(kp1 + 32); a11.q[1] = *(const uint4*)(kp1 + 48);

      v8f s0 = {0,0,0,0,0,0,0,0}, s1 = {0,0,0,0,0,0,0,0};
      s0 = __builtin_amdgcn_wmma_f32_16x16x32_bf16(false, a00.v, false, qb0.v, (short)0, s0, false, false);
      s0 = __builtin_amdgcn_wmma_f32_16x16x32_bf16(false, a01.v, false, qb1.v, (short)0, s0, false, false);
      s1 = __builtin_amdgcn_wmma_f32_16x16x32_bf16(false, a10.v, false, qb0.v, (short)0, s1, false, false);
      s1 = __builtin_amdgcn_wmma_f32_16x16x32_bf16(false, a11.v, false, qb1.v, (short)0, s1, false, false);

      // online softmax over the 32 m values of this tile (per query column n)
      float sv[16];
      #pragma unroll
      for (int i = 0; i < 8; ++i) { sv[i] = s0[i]; sv[8 + i] = s1[i]; }
      float tmax = sv[0];
      #pragma unroll
      for (int i = 1; i < 16; ++i) tmax = fmaxf(tmax, sv[i]);
      tmax = fmaxf(tmax, __shfl_xor(tmax, 16, 32));  // lanes L/L+16 share column n
      float nmax = fmaxf(rmax, tmax);
      float al   = __expf(rmax - nmax);
      float ps = 0.0f;
      #pragma unroll
      for (int i = 0; i < 16; ++i) { float p = __expf(sv[i] - nmax); sv[i] = p; ps += p; }
      ps += __shfl_xor(ps, 16, 32);
      rsum = rsum * al + ps;
      rmax = nmax;
      if (lane < 16) sh_alpha[nloc] = al;
      #pragma unroll
      for (int i = 0; i < 8; ++i) {
        sh_p[nloc * 32 + halfA + i]      = f2bf(sv[i]);        // m = halfA+i
        sh_p[nloc * 32 + 16 + halfA + i] = f2bf(sv[8 + i]);    // m = 16+halfA+i
      }
    }
    __syncthreads();

    // --- all waves: rescale O, accumulate P @ V ---
    float4 alo = *(const float4*)&sh_alpha[halfA];
    float4 ahi = *(const float4*)&sh_alpha[halfA + 4];
    float av[8] = {alo.x, alo.y, alo.z, alo.w, ahi.x, ahi.y, ahi.z, ahi.w};
    #pragma unroll
    for (int i = 0; i < 8; ++i) {
      o0[i] *= av[i]; o1[i] *= av[i]; o2[i] *= av[i]; o3[i] *= av[i];
    }

    Frag16 pa;                                        // P A-fragment
    const unsigned short* pp = &sh_p[nloc * 32 + halfA];
    pa.q[0] = *(const uint4*)(pp);
    pa.q[1] = *(const uint4*)(pp + 16);

    #pragma unroll
    for (int j = 0; j < 4; ++j) {
      int vcol = (wid * 4 + j) * 16 + nloc;
      Frag16 vbf;                                     // V B-fragment from LDS
      const unsigned short* vp = &sh_v[vcol * 32 + halfA];
      vbf.q[0] = *(const uint4*)(vp);
      vbf.q[1] = *(const uint4*)(vp + 16);
      v8f& oo = (j == 0) ? o0 : (j == 1) ? o1 : (j == 2) ? o2 : o3;
      oo = __builtin_amdgcn_wmma_f32_16x16x32_bf16(false, pa.v, false, vbf.v, (short)0, oo, false, false);
    }
    __syncthreads();                                  // protect sh_p / sh_v reuse
  }

  // --- normalize by softmax denominator and store O ---
  if (wid == 0 && lane < 16) sh_l[nloc] = rsum;
  __syncthreads();
  float4 llo = *(const float4*)&sh_l[halfA];
  float4 lhi = *(const float4*)&sh_l[halfA + 4];
  float li[8] = {1.f/llo.x, 1.f/llo.y, 1.f/llo.z, 1.f/llo.w,
                 1.f/lhi.x, 1.f/lhi.y, 1.f/lhi.z, 1.f/lhi.w};
  #pragma unroll
  for (int j = 0; j < 4; ++j) {
    v8f& oo = (j == 0) ? o0 : (j == 1) ? o1 : (j == 2) ? o2 : o3;
    int vcol = (wid * 4 + j) * 16 + nloc;
    #pragma unroll
    for (int i = 0; i < 8; ++i) {
      int nrow = qt * 16 + halfA + i;
      ow[((size_t)(b * NN) + nrow) * VV + vcol] = oo[i] * li[i];
    }
  }
}

// ---------------------------------------------------------------------------
// Kernel 3: e = relu(BN(We @ out)) + x  -> d_out  [B][C][N] f32
// ---------------------------------------------------------------------------
__global__ __launch_bounds__(256) void pam_out_proj(
    const float* __restrict__ ow,                    // [B][N][V]
    const float* __restrict__ We,                    // [C][V]
    const float* __restrict__ ge, const float* __restrict__ be,
    const float* __restrict__ me, const float* __restrict__ ve,
    const float* __restrict__ x,
    float* __restrict__ out)
{
  __shared__ float os[32 * VV];                      // out tile [n][v], 32 KB
  const int blk = blockIdx.x;
  const int b   = blk / (NN / 32);
  const int n0  = (blk % (NN / 32)) * 32;
  const int t   = threadIdx.x;

  #pragma unroll 4
  for (int r = 0; r < 32; ++r) {
    int idx = t + 256 * r;                           // n = idx>>8, v = idx&255
    os[idx] = ow[((size_t)(b * NN) + n0 + (idx >> 8)) * VV + (idx & 255)];
  }
  __syncthreads();

  const int n  = t & 31;
  const int cg = t >> 5;
  for (int g = 0; g < 8; ++g) {
    const int c0 = cg * 64 + g * 8;
    float acc[8] = {0.f,0.f,0.f,0.f,0.f,0.f,0.f,0.f};
    const float* wr = We + (size_t)c0 * VV;
    for (int v = 0; v < VV; ++v) {
      float ovv = os[n * VV + v];
      #pragma unroll
      for (int u = 0; u < 8; ++u) acc[u] += ovv * wr[(size_t)u * VV + v];
    }
    #pragma unroll
    for (int u = 0; u < 8; ++u) {
      int c = c0 + u;
      float s  = ge[c] * rsqrtf(ve[c] + EPSF);
      float sh = be[c] - me[c] * s;
      float y  = fmaxf(acc[u] * s + sh, 0.0f);
      size_t gi = ((size_t)b * CC + c) * NN + n0 + n;
      out[gi] = y + x[gi];
    }
  }
}

// ---------------------------------------------------------------------------
extern "C" void kernel_launch(void* const* d_in, const int* in_sizes, int n_in,
                              void* d_out, int out_size, void* d_ws, size_t ws_size,
                              hipStream_t stream) {
  const float* x  = (const float*)d_in[0];
  const float* Wq = (const float*)d_in[1];
  const float* gq = (const float*)d_in[2];
  const float* bq = (const float*)d_in[3];
  const float* mq = (const float*)d_in[4];
  const float* vq = (const float*)d_in[5];
  const float* Wk = (const float*)d_in[6];
  const float* gk = (const float*)d_in[7];
  const float* bk = (const float*)d_in[8];
  const float* mk = (const float*)d_in[9];
  const float* vk = (const float*)d_in[10];
  const float* Wv = (const float*)d_in[11];
  const float* bv = (const float*)d_in[12];
  const float* We = (const float*)d_in[13];
  const float* ge = (const float*)d_in[14];
  const float* be = (const float*)d_in[15];
  const float* me = (const float*)d_in[16];
  const float* ve = (const float*)d_in[17];
  float* out = (float*)d_out;

  char* w = (char*)d_ws;
  unsigned short* qws = (unsigned short*)w; w += (size_t)BB * NN * KK * 2;   // 2.36 MB
  unsigned short* kws = (unsigned short*)w; w += (size_t)BB * NN * KK * 2;   // 2.36 MB
  unsigned short* vws = (unsigned short*)w; w += (size_t)BB * NN * VV * 2;   // 9.44 MB
  float*          ows = (float*)w;                                           // 18.9 MB

  pam_proj_qkv<<<BB * (NN / 32), 256, 0, stream>>>(
      x, Wq, gq, bq, mq, vq, Wk, gk, bk, mk, vk, Wv, bv, qws, kws, vws);
  pam_attn_kernel<<<BB * (NN / 16), 128, 0, stream>>>(qws, kws, vws, ows);
  pam_out_proj<<<BB * (NN / 32), 256, 0, stream>>>(
      ows, We, ge, be, me, ve, x, out);
}
